// QuaternionTransformer_34213709480406
// MI455X (gfx1250) — compile-verified, matrix operand order checked by reading, not run on hardware
//
#include <hip/hip_runtime.h>

// ============================================================================
// Quaternion Transformer decoder forward for gfx1250 (MI455X).
//
// All GEMMs: single NT bf16 WMMA kernel (v_wmma_f32_16x16x32_bf16, fp32 acc),
// double-buffered LDS staging via GLOBAL_LOAD_ASYNC_TO_LDS_B128 (ASYNCcnt).
// Block tile 128x128x32, 8 waves (4Mx2N), each wave 32x64 = 8 WMMA per K-step.
// Operands are pre-converted to bf16 exactly once by producer epilogues.
// qlinear = real GEMM against the expanded signed-permutation weight (bf16).
// probs x V is made NT by writing the V projection bf16-transposed.
//
// B=4, T=S=512, D=512, H=8, HID=2048, 4 layers; every GEMM dim divides the
// tile sizes -> no edge guards. Workspace arena ~280MB carved from d_ws.
// ============================================================================

typedef __attribute__((ext_vector_type(16))) __bf16 v16bf;
typedef __attribute__((ext_vector_type(8)))  float  v8f;

#define BLK_M 128
#define BLK_N 128
#define BLK_K 32
#define LDS_S 40                    // ushorts per LDS row (80B: 16B-aligned rows)
#define A_LDS_ELEMS (BLK_M * LDS_S) // 5120 ushorts
#define B_LDS_ELEMS (BLK_N * LDS_S) // 5120 ushorts
#define BUF_ELEMS   (A_LDS_ELEMS + B_LDS_ELEMS) // 10240 ushorts per buffer

union FragU { unsigned int u[8]; v16bf v; };

__device__ __forceinline__ unsigned short f32_to_bf16(float f) {
    unsigned int u = __builtin_bit_cast(unsigned int, f);
    u = (u + 0x7FFFu + ((u >> 16) & 1u)) >> 16;   // round-to-nearest-even
    return (unsigned short)u;
}

// Flat addresses in the LDS aperture keep the wave-relative LDS byte offset in
// addr[31:0] (ISA 10.2 aperture mapping), so truncation yields the DS address.
__device__ __forceinline__ unsigned lds_addr_of(const void* p) {
    return (unsigned)(unsigned long long)p;
}

// GLOBAL_LOAD_ASYNC_TO_LDS_B128: per-lane 16B global -> LDS copy, ASYNCcnt.
__device__ __forceinline__ void async_copy_b128(const unsigned short* g, unsigned lds) {
    asm volatile("global_load_async_to_lds_b128 %0, %1, off"
                 :: "v"(lds), "v"(g) : "memory");
}

// Stage one BLK_M x BLK_K A tile + BLK_N x BLK_K B tile (bf16) asynchronously.
// 4 async instructions per thread: 2 for A, 2 for B (512 chunks each / 256thr).
__device__ __forceinline__ void stage_tile_async(
    const unsigned short* __restrict__ Ab, const unsigned short* __restrict__ Bb,
    long lda, long ldb, int blockM, int blockN, int k0,
    unsigned ldsA, unsigned ldsB, int tid)
{
    int c = tid, r = c >> 2, q = c & 3;
    async_copy_b128(Ab + (long)(blockM + r) * lda + (k0 + q * 8),
                    ldsA + (unsigned)(r * LDS_S + q * 8) * 2);
    async_copy_b128(Bb + (long)(blockN + r) * ldb + (k0 + q * 8),
                    ldsB + (unsigned)(r * LDS_S + q * 8) * 2);
    c = tid + 256; r = c >> 2; q = c & 3;
    async_copy_b128(Ab + (long)(blockM + r) * lda + (k0 + q * 8),
                    ldsA + (unsigned)(r * LDS_S + q * 8) * 2);
    async_copy_b128(Bb + (long)(blockN + r) * ldb + (k0 + q * 8),
                    ldsB + (unsigned)(r * LDS_S + q * 8) * 2);
}

// ----------------------------------------------------------------------------
// GEMM (NT): C[M,N] = A[M,K] * B[N,K]^T, A/B bf16 row-major, fp32 accumulate.
// Epilogue writes any of: Cf (fp32, +bias), Cb (bf16), Cbt (bf16 transposed).
// Batched via blockIdx.z with split (outer, head) strides.
// ----------------------------------------------------------------------------
__global__ __launch_bounds__(256) void gemm_wmma_async(
    const unsigned short* __restrict__ A, const unsigned short* __restrict__ Bm,
    float* __restrict__ Cf, unsigned short* __restrict__ Cb,
    unsigned short* __restrict__ Cbt, const float* __restrict__ bias,
    int K, long lda, long ldb, long ldc, long ldct, int nH,
    long aSB, long aSH, long bSB, long bSH, long cSB, long cSH)
{
    __shared__ __align__(16) unsigned short smem[2 * BUF_ELEMS];

    const int z  = blockIdx.z;
    const int zb = z / nH, zh = z % nH;
    A  += (long)zb * aSB + (long)zh * aSH;
    Bm += (long)zb * bSB + (long)zh * bSH;

    const int tid   = threadIdx.x;
    const int lane  = tid & 31;
    const int wid   = tid >> 5;
    const int waveM = wid & 3;   // 4 x 32 rows
    const int waveN = wid >> 2;  // 2 x 64 cols

    const int blockM = blockIdx.y * BLK_M;
    const int blockN = blockIdx.x * BLK_N;

    const unsigned ldsBase = lds_addr_of(smem);
    const unsigned ldsA0 = ldsBase;
    const unsigned ldsB0 = ldsBase + A_LDS_ELEMS * 2;
    const unsigned ldsA1 = ldsBase + BUF_ELEMS * 2;
    const unsigned ldsB1 = ldsA1 + A_LDS_ELEMS * 2;

    v8f acc[2][4];
#pragma unroll
    for (int i = 0; i < 2; ++i)
#pragma unroll
        for (int j = 0; j < 4; ++j)
#pragma unroll
            for (int e = 0; e < 8; ++e) acc[i][j][e] = 0.0f;

    const int rowLane = lane & 15;
    const int kbase   = (lane >> 4) << 3;   // 0 or 8 per documented bf16 layout

    const int nT = K / BLK_K;
    stage_tile_async(A, Bm, lda, ldb, blockM, blockN, 0, ldsA0, ldsB0, tid);

    for (int t = 0; t < nT; ++t) {
        const int cur = t & 1;
        if (t + 1 < nT) {
            stage_tile_async(A, Bm, lda, ldb, blockM, blockN, (t + 1) * BLK_K,
                             cur ? ldsA0 : ldsA1, cur ? ldsB0 : ldsB1, tid);
            asm volatile("s_wait_asynccnt 0x4" ::: "memory");  // next tile may fly
        } else {
            asm volatile("s_wait_asynccnt 0x0" ::: "memory");
        }
        __syncthreads();   // all waves' copies for buffer `cur` have landed

        const int aOff = cur * BUF_ELEMS;
        const int bOff = aOff + A_LDS_ELEMS;

        v16bf afrag[2], bfrag[4];
#pragma unroll
        for (int i = 0; i < 2; ++i) {
            const int m = waveM * 32 + i * 16 + rowLane;
            FragU fu;
#pragma unroll
            for (int r = 0; r < 8; ++r) {
                const int kk = (r < 4) ? (kbase + r * 2) : (16 + kbase + (r - 4) * 2);
                fu.u[r] = *(const unsigned int*)&smem[aOff + m * LDS_S + kk];
            }
            afrag[i] = fu.v;
        }
#pragma unroll
        for (int j = 0; j < 4; ++j) {
            const int n = waveN * 64 + j * 16 + rowLane;
            FragU fu;
#pragma unroll
            for (int r = 0; r < 8; ++r) {
                const int kk = (r < 4) ? (kbase + r * 2) : (16 + kbase + (r - 4) * 2);
                fu.u[r] = *(const unsigned int*)&smem[bOff + n * LDS_S + kk];
            }
            bfrag[j] = fu.v;
        }

#pragma unroll
        for (int i = 0; i < 2; ++i)
#pragma unroll
            for (int j = 0; j < 4; ++j)
                acc[i][j] = __builtin_amdgcn_wmma_f32_16x16x32_bf16(
                    false, afrag[i], false, bfrag[j], (short)0, acc[i][j], false, false);

        __syncthreads();   // reads done before next iteration overwrites buffer
    }

    // ---- epilogue: documented 16x16 f32 D layout ----
    const int nLane = lane & 15;
    const int mOff  = (lane >> 4) * 8;
    float* CfB = Cf ? Cf + (long)zb * cSB + (long)zh * cSH : nullptr;
    unsigned short* CbB = Cb ? Cb + (long)zb * cSB + (long)zh * cSH : nullptr;
#pragma unroll
    for (int i = 0; i < 2; ++i)
#pragma unroll
        for (int j = 0; j < 4; ++j) {
            const int colG = blockN + waveN * 64 + j * 16 + nLane;
            const float bv = bias ? bias[colG] : 0.0f;
#pragma unroll
            for (int v = 0; v < 8; ++v) {
                const int rowG = blockM + waveM * 32 + i * 16 + v + mOff;
                const float val = acc[i][j][v] + bv;
                if (CfB) CfB[(long)rowG * ldc + colG] = val;
                if (CbB) CbB[(long)rowG * ldc + colG] = f32_to_bf16(val);
                if (Cbt) Cbt[(long)colG * ldct + rowG] = f32_to_bf16(val);
            }
        }
}

// ----------------------------------------------------------------------------
// Expand quaternion weight into the real signed-permutation matrix (bf16):
// Wexp[(o*4+c)][(i*4+q)] = sgn(q,c) * W_{wsel(q,c)}[o][i]
// ----------------------------------------------------------------------------
__global__ __launch_bounds__(256) void expand_qweight(
    const float* __restrict__ Wr, const float* __restrict__ Wi,
    const float* __restrict__ Wj, const float* __restrict__ Wk,
    unsigned short* __restrict__ Wexp, int outF, int inF)
{
    const int idx = blockIdx.x * 256 + threadIdx.x;
    if (idx >= outF * inF) return;
    const int o = idx / inF, i = idx % inF;
    const float w[4] = { Wr[idx], Wi[idx], Wj[idx], Wk[idx] };
    const int   ws[4][4] = { {0,1,2,3}, {1,0,3,2}, {2,3,0,1}, {3,2,1,0} };
    const float sg[4][4] = { {1,1,1,1}, {-1,1,-1,1}, {-1,1,1,-1}, {-1,-1,1,1} };
    const long K4 = 4L * inF;
#pragma unroll
    for (int c = 0; c < 4; ++c) {
        const unsigned short h0 = f32_to_bf16(sg[0][c] * w[ws[0][c]]);
        const unsigned short h1 = f32_to_bf16(sg[1][c] * w[ws[1][c]]);
        const unsigned short h2 = f32_to_bf16(sg[2][c] * w[ws[2][c]]);
        const unsigned short h3 = f32_to_bf16(sg[3][c] * w[ws[3][c]]);
        uint2 pk;
        pk.x = (unsigned)h0 | ((unsigned)h1 << 16);
        pk.y = (unsigned)h2 | ((unsigned)h3 << 16);
        *(uint2*)(Wexp + (long)(o * 4 + c) * K4 + (long)i * 4) = pk;
    }
}

// ----------------------------------------------------------------------------
__global__ __launch_bounds__(256) void f32_to_bf16_kernel(
    const float* __restrict__ in, unsigned short* __restrict__ out, long n4)
{
    const long idx = (long)blockIdx.x * 256 + threadIdx.x;  // one float4 each
    if (idx >= n4) return;
    const float4 v = *(const float4*)(in + idx * 4);
    uint2 pk;
    pk.x = (unsigned)f32_to_bf16(v.x) | ((unsigned)f32_to_bf16(v.y) << 16);
    pk.y = (unsigned)f32_to_bf16(v.z) | ((unsigned)f32_to_bf16(v.w) << 16);
    *(uint2*)(out + idx * 4) = pk;
}

// ----------------------------------------------------------------------------
// Embedding gather + angular scaling; writes fp32 master + bf16 copy.
// ----------------------------------------------------------------------------
__global__ __launch_bounds__(256) void embed_rope_kernel(
    const int* __restrict__ ids,
    const float* __restrict__ er, const float* __restrict__ ei,
    const float* __restrict__ ej, const float* __restrict__ ek,
    float* __restrict__ x, unsigned short* __restrict__ xb, int B, int T, int D)
{
    const int idx = blockIdx.x * 256 + threadIdx.x;
    if (idx >= B * T * D) return;
    const int d  = idx % D;
    const int bt = idx / D;
    const int t  = bt % T;
    const int id = ids[bt];
    const int j  = (d < D / 2) ? d : d - D / 2;
    const float e    = (2.0f * (float)j) / (float)D + 1e-6f;
    const float invf = __expf(-e * 9.2103403719761836f);   // ln(10000)
    const float ang  = (float)t * invf;
    const float a    = (d < D / 2) ? sinf(ang) : cosf(ang);
    const long  o    = (long)idx * 4;
    const long  eo   = (long)id * D + d;
    const float v0 = er[eo] * a, v1 = ei[eo] * a, v2 = ej[eo] * a, v3 = ek[eo] * a;
    x[o + 0] = v0; x[o + 1] = v1; x[o + 2] = v2; x[o + 3] = v3;
    uint2 pk;
    pk.x = (unsigned)f32_to_bf16(v0) | ((unsigned)f32_to_bf16(v1) << 16);
    pk.y = (unsigned)f32_to_bf16(v2) | ((unsigned)f32_to_bf16(v3) << 16);
    *(uint2*)(xb + o) = pk;
}

// ----------------------------------------------------------------------------
// Masked, scaled softmax over the key axis (S == 512); bf16 probs out.
// Reference order: mask -> -inf, then scale, then softmax.
// ----------------------------------------------------------------------------
__global__ __launch_bounds__(256) void softmax_kernel(
    const float* __restrict__ sc, unsigned short* __restrict__ pb,
    const int* __restrict__ mask, int H, int T, int S, float scale)
{
    const long row = blockIdx.x;                 // (b*H + h)*T + t
    const float* p = sc + row * (long)S;
    unsigned short* po = pb + row * (long)S;
    const int tid = threadIdx.x;
    const int b   = (int)(row / ((long)H * (long)T));
    float x0 = p[tid], x1 = p[tid + 256];
    if (mask) {
        const int* mr = mask + (long)b * S;
        x0 = (mr[tid]       == 0) ? -__builtin_inff() : x0 * scale;
        x1 = (mr[tid + 256] == 0) ? -__builtin_inff() : x1 * scale;
    } else { x0 *= scale; x1 *= scale; }

    float m = fmaxf(x0, x1);
#pragma unroll
    for (int off = 16; off > 0; off >>= 1) m = fmaxf(m, __shfl_xor(m, off, 32));
    __shared__ float smax[8], ssum[8];
    const int lane = tid & 31, w = tid >> 5;
    if (lane == 0) smax[w] = m;
    __syncthreads();
    m = smax[0];
#pragma unroll
    for (int i = 1; i < 8; ++i) m = fmaxf(m, smax[i]);

    const float e0 = __expf(x0 - m), e1 = __expf(x1 - m);
    float s = e0 + e1;
#pragma unroll
    for (int off = 16; off > 0; off >>= 1) s += __shfl_xor(s, off, 32);
    if (lane == 0) ssum[w] = s;
    __syncthreads();
    s = 0.0f;
#pragma unroll
    for (int i = 0; i < 8; ++i) s += ssum[i];
    const float inv = 1.0f / s;
    po[tid]       = f32_to_bf16(e0 * inv);
    po[tid + 256] = f32_to_bf16(e1 * inv);
}

// ----------------------------------------------------------------------------
// modReLU: reads fp32 hidden, writes bf16 (only consumer is the fc2 GEMM).
// ----------------------------------------------------------------------------
__global__ __launch_bounds__(256) void modrelu_kernel(
    const float* __restrict__ h, unsigned short* __restrict__ hb,
    const float* __restrict__ bias, int total, int F)
{
    const int idx = blockIdx.x * 256 + threadIdx.x;
    if (idx >= total) return;
    const int f = idx % F;
    const float4 v = *(const float4*)(h + (long)idx * 4);
    const float n = sqrtf(v.x*v.x + v.y*v.y + v.z*v.z + v.w*v.w);
    const float z = n + bias[f];
    const float g = 0.5f * z * (1.0f + erff(z * 0.70710678118654752f));
    const float s = g / (n + 1e-6f);
    uint2 pk;
    pk.x = (unsigned)f32_to_bf16(v.x * s) | ((unsigned)f32_to_bf16(v.y * s) << 16);
    pk.y = (unsigned)f32_to_bf16(v.z * s) | ((unsigned)f32_to_bf16(v.w * s) << 16);
    *(uint2*)(hb + (long)idx * 4) = pk;
}

// ----------------------------------------------------------------------------
// xout = qnorm(xin + delta); writes fp32 master + bf16 copy.
// ----------------------------------------------------------------------------
__global__ __launch_bounds__(256) void add_qnorm_kernel(
    const float* __restrict__ xin, const float* __restrict__ delta,
    const float* __restrict__ gamma, const float* __restrict__ beta,
    float* __restrict__ xout, unsigned short* __restrict__ xoutb, int total, int D)
{
    const int idx = blockIdx.x * 256 + threadIdx.x;
    if (idx >= total) return;
    const int d = idx % D;
    const float4 a = *(const float4*)(xin   + (long)idx * 4);
    const float4 b = *(const float4*)(delta + (long)idx * 4);
    float4 v; v.x = a.x + b.x; v.y = a.y + b.y; v.z = a.z + b.z; v.w = a.w + b.w;
    const float n   = sqrtf(v.x*v.x + v.y*v.y + v.z*v.z + v.w*v.w + 1e-6f);
    const float inv = 1.0f / (n + 1e-6f);
    const float4 g  = *(const float4*)(gamma + (long)d * 4);
    const float4 be = *(const float4*)(beta  + (long)d * 4);
    float4 o;
    o.x = g.x * v.x * inv + be.x; o.y = g.y * v.y * inv + be.y;
    o.z = g.z * v.z * inv + be.z; o.w = g.w * v.w * inv + be.w;
    *(float4*)(xout + (long)idx * 4) = o;
    uint2 pk;
    pk.x = (unsigned)f32_to_bf16(o.x) | ((unsigned)f32_to_bf16(o.y) << 16);
    pk.y = (unsigned)f32_to_bf16(o.z) | ((unsigned)f32_to_bf16(o.w) << 16);
    *(uint2*)(xoutb + (long)idx * 4) = pk;
}

// ============================================================================
// Host orchestration
// ============================================================================

static void run_gemm(const unsigned short* A, const unsigned short* Bm,
                     float* Cf, unsigned short* Cb, unsigned short* Cbt,
                     const float* bias, int M, int N, int K,
                     long lda, long ldb, long ldc, long ldct,
                     int batch, int nH,
                     long aSB, long aSH, long bSB, long bSH, long cSB, long cSH,
                     hipStream_t stream)
{
    dim3 grid(N / BLK_N, M / BLK_M, batch);
    gemm_wmma_async<<<grid, 256, 0, stream>>>(A, Bm, Cf, Cb, Cbt, bias, K,
                                              lda, ldb, ldc, ldct, nH,
                                              aSB, aSH, bSB, bSH, cSB, cSH);
}

extern "C" void kernel_launch(void* const* d_in, const int* in_sizes, int n_in,
                              void* d_out, int out_size, void* d_ws, size_t ws_size,
                              hipStream_t stream)
{
    constexpr int B = 4, T = 512, S = 512, D = 512, H = 8, HID = 2048;
    constexpr int M = B * T;                        // 2048
    constexpr long XN  = (long)B * T * D * 4;       // 4,194,304
    constexpr long SCN = (long)B * H * T * S;       // 8,388,608
    constexpr long H1N = (long)B * T * HID * 4;     // 16,777,216
    constexpr long WEN = (long)(4 * HID) * (4 * D); // 16,777,216 (max Wexp)
    (void)ws_size; (void)in_sizes; (void)n_in;

    // ---- workspace arena ----
    char* wp = (char*)d_ws;
    auto allocF = [&](long n) { float* p = (float*)wp; wp += n * 4; return p; };
    auto allocH = [&](long n) { unsigned short* p = (unsigned short*)wp; wp += n * 2; return p; };
    float* X0    = allocF(XN);
    float* X1    = allocF(XN);
    float* DELTA = allocF(XN);
    float* SC    = allocF(SCN);
    float* H1    = allocF(H1N);
    unsigned short* X0b  = allocH(XN);
    unsigned short* X1b  = allocH(XN);
    unsigned short* QLb  = allocH(XN);
    unsigned short* KLb  = allocH(XN);
    unsigned short* VLt  = allocH(XN);   // V projection, bf16 TRANSPOSED [4D][M]
    unsigned short* ATTb = allocH(XN);
    unsigned short* MEMb = allocH(XN);
    unsigned short* SCb  = allocH(SCN);
    unsigned short* H1b  = allocH(H1N);
    unsigned short* WEXPb = allocH(WEN);

    // ---- input pointer map (setup_inputs dict order, leaves depth-first) ----
    // 0 input_ids, 1 tgt_mask, 2 memory, 3..6 embed r,i,j,k; layer base 7+57l:
    // +0..4 sa.q{r,i,j,k,b} +5..9 sa.k +10..14 sa.v +15..19 sa.o
    // +20..39 cross{q,k,v,o} +40..44 fc1 +45..49 fc2 +50 mod_bias
    // +51 g1 +52 b1 +53 g2 +54 b2 +55 g3 +56 b3
    auto P = [&](int i) { return (const float*)d_in[i]; };
    const int*   ids   = (const int*)d_in[0];
    const int*   tmask = (const int*)d_in[1];
    const float* mem   = (const float*)d_in[2];

    // qlinear: out[M, 4*outF] = A[M, 4*inF] x Wexp^T (+bias) via bf16 WMMA.
    auto qlinear = [&](const unsigned short* Ab, int pb, int inF, int outF,
                       float* Cf, unsigned short* Cb, unsigned short* Cbt) {
        const int tot = outF * inF;
        expand_qweight<<<(tot + 255) / 256, 256, 0, stream>>>(
            P(pb + 0), P(pb + 1), P(pb + 2), P(pb + 3), WEXPb, outF, inF);
        run_gemm(Ab, WEXPb, Cf, Cb, Cbt, P(pb + 4),
                 M, 4 * outF, 4 * inF,
                 4L * inF, 4L * inF, 4L * outF, (long)M,
                 1, 1, 0, 0, 0, 0, 0, 0, stream);
    };

    const float scale = 1.0f / sqrtf((float)(64 * 4) + 2e-6f);

    // Multi-head quaternion attention; result delta (fp32) left in DELTA.
    auto attention = [&](const unsigned short* xinb, const unsigned short* kvb,
                         int pb, const int* mask) {
        qlinear(xinb, pb + 0,  D, D, nullptr, QLb, nullptr);
        qlinear(kvb,  pb + 5,  D, D, nullptr, KLb, nullptr);
        qlinear(kvb,  pb + 10, D, D, nullptr, nullptr, VLt);
        // scores[b,h,t,s] = Qh[t,:256] . Kh[s,:256]  (NT, batched over b*h)
        run_gemm(QLb, KLb, SC, nullptr, nullptr, nullptr,
                 T, S, 256,
                 2048, 2048, S, 0,
                 B * H, H,
                 (long)T * 2048, 256,
                 (long)T * 2048, 256,
                 (long)H * T * S, (long)T * S, stream);
        softmax_kernel<<<B * H * T, 256, 0, stream>>>(SC, SCb, mask, H, T, S, scale);
        // out[b,h,t,f] = sum_s P[t,s] * Vt[f,s]  (NT thanks to transposed V)
        run_gemm(SCb, VLt, nullptr, ATTb, nullptr, nullptr,
                 T, 256, S,
                 S, (long)M, 2048, 0,
                 B * H, H,
                 (long)H * T * S, (long)T * S,
                 (long)T, (long)256 * M,
                 (long)T * 2048, 256, stream);
        qlinear(ATTb, pb + 15, D, D, DELTA, nullptr, nullptr);
    };

    // ---- memory -> bf16 (once), embed + rope ----
    f32_to_bf16_kernel<<<(int)(XN / 4 / 256), 256, 0, stream>>>(mem, MEMb, XN / 4);
    embed_rope_kernel<<<(B * T * D + 255) / 256, 256, 0, stream>>>(
        ids, P(3), P(4), P(5), P(6), X0, X0b, B, T, D);

    float* cur = X0;  unsigned short* curb = X0b;
    float* nxt = X1;  unsigned short* nxtb = X1b;
    const int nBTD = B * T * D;

    for (int l = 0; l < 4; ++l) {
        const int base = 7 + l * 57;

        attention(curb, curb, base + 0, tmask);
        add_qnorm_kernel<<<nBTD / 256, 256, 0, stream>>>(
            cur, DELTA, P(base + 51), P(base + 52), nxt, nxtb, nBTD, D);
        { float* t = cur; cur = nxt; nxt = t; }
        { unsigned short* t = curb; curb = nxtb; nxtb = t; }

        attention(curb, MEMb, base + 20, nullptr);
        add_qnorm_kernel<<<nBTD / 256, 256, 0, stream>>>(
            cur, DELTA, P(base + 53), P(base + 54), nxt, nxtb, nBTD, D);
        { float* t = cur; cur = nxt; nxt = t; }
        { unsigned short* t = curb; curb = nxtb; nxtb = t; }

        qlinear(curb, base + 40, D, HID, H1, nullptr, nullptr);
        modrelu_kernel<<<(B * T * HID) / 256, 256, 0, stream>>>(
            H1, H1b, P(base + 50), B * T * HID, HID);
        qlinear(H1b, base + 45, HID, D, DELTA, nullptr, nullptr);
        add_qnorm_kernel<<<nBTD / 256, 256, 0, stream>>>(
            cur, DELTA, P(base + 55), P(base + 56), nxt, nxtb, nBTD, D);
        { float* t = cur; cur = nxt; nxt = t; }
        { unsigned short* t = curb; curb = nxtb; nxtb = t; }
    }

    hipMemcpyAsync(d_out, cur, (size_t)out_size * sizeof(float),
                   hipMemcpyDeviceToDevice, stream);
}
